// PointNetSetAbstraction_62062277427545
// MI455X (gfx1250) — matrix operand order repaired and checked by reading, not compiled, not run on hardware
//
#include <hip/hip_runtime.h>

typedef __attribute__((ext_vector_type(16))) _Float16 v16h;
typedef __attribute__((ext_vector_type(8)))  float    v8f;

#define BATCH    8
#define NPTS     8192
#define NPOINT_  2048
#define NSAMPLE_ 32
#define DFEAT    32
#define KDIM     64                            // padded input-channel dim (layer inputs)
#define MROWS    (BATCH * NPOINT_ * NSAMPLE_)  // 524288 rows through the MLP
#define BN_EPS_  1e-5f

// Fragment-order layouts (all 16-bit WMMA shapes, wave32):
//  A (16x32 f16, ISA 7.12.2): lane = half*16+ml holds row ml; element i -> K as below.
//  B (32x16 f16):             lane = half*16+nl holds col nl; element i -> K = afrag_K(half, ks, i), same as A.
//  Activations are stored pre-packed per-lane: Afrag[((mtile*(C/32)+ks)*32+lane)*16+i],
//  so every WMMA operand load is one contiguous 32B (2x b128) coalesced load.
__device__ __forceinline__ int afrag_K(int half, int ks, int i) {
    return ks * 32 + ((i >> 3) << 4) + half * 8 + (i & 7);
}

// ---------------------------------------------------------------------------
// Prep: weights -> f16 in B-fragment order; zero BN accumulators.
// ---------------------------------------------------------------------------
__global__ void prep_kernel(const float* __restrict__ w1, const float* __restrict__ w2,
                            const float* __restrict__ w3,
                            _Float16* __restrict__ wf1, _Float16* __restrict__ wf2,
                            _Float16* __restrict__ wf3, float* __restrict__ stats) {
    int t = blockIdx.x * blockDim.x + threadIdx.x;      // 8192 threads
    if (t < 512) stats[t] = 0.0f;
    if (t < 4096) {                                     // layers 1,2: NOUT=64 -> 4*2*32*16
        int i = t & 15, lane = (t >> 4) & 31, ks = (t >> 9) & 1, nt = t >> 10;
        int K = ks * 32 + ((i >> 3) << 4) + (lane >> 4) * 8 + (i & 7);
        int n = nt * 16 + (lane & 15);
        wf1[t] = (K < 35) ? (_Float16)w1[n * 35 + K] : (_Float16)0.0f;   // w1:[64,35]
        wf2[t] = (_Float16)w2[n * 64 + K];                               // w2:[64,64]
    }
    if (t < 8192) {                                     // layer 3: NOUT=128 -> 8*2*32*16
        int i = t & 15, lane = (t >> 4) & 31, ks = (t >> 9) & 1, nt = t >> 10;
        int K = ks * 32 + ((i >> 3) << 4) + (lane >> 4) * 8 + (i & 7);
        int n = nt * 16 + (lane & 15);
        wf3[t] = (_Float16)w3[n * 64 + K];                               // w3:[128,64]
    }
}

// ---------------------------------------------------------------------------
// Farthest point sampling: one block per batch, dist[] in LDS, serial loop
// with block-tree argmax (tie -> lowest index). Also writes new_xyz output.
// ---------------------------------------------------------------------------
__global__ void fps_kernel(const float* __restrict__ xyz, int* __restrict__ fps_idx,
                           float* __restrict__ new_xyz) {
    __shared__ float dist[NPTS];        // 32 KB
    __shared__ float rv[512];
    __shared__ int   ri[512];
    __shared__ int   sel[NPOINT_];      // 8 KB
    const int b   = blockIdx.x;
    const int tid = threadIdx.x;
    const float* xb = xyz + (size_t)b * NPTS * 3;

    for (int i = tid; i < NPTS; i += 512) dist[i] = 1e10f;
    if (tid == 0) sel[0] = 0;
    __syncthreads();

    int last = 0;
    for (int it = 1; it < NPOINT_; ++it) {
        const float px = xb[last * 3 + 0];
        const float py = xb[last * 3 + 1];
        const float pz = xb[last * 3 + 2];
        float bv = -1.0f; int bi = 0;
        for (int i = tid; i < NPTS; i += 512) {
            float dx = xb[i * 3 + 0] - px;
            float dy = xb[i * 3 + 1] - py;
            float dz = xb[i * 3 + 2] - pz;
            float d  = dx * dx + dy * dy + dz * dz;
            float dd = fminf(dist[i], d);
            dist[i] = dd;
            if (dd > bv) { bv = dd; bi = i; }
        }
        rv[tid] = bv; ri[tid] = bi;
        __syncthreads();
        for (int s = 256; s > 0; s >>= 1) {
            if (tid < s) {
                float ov = rv[tid + s]; int oi = ri[tid + s];
                if (ov > rv[tid] || (ov == rv[tid] && oi < ri[tid])) { rv[tid] = ov; ri[tid] = oi; }
            }
            __syncthreads();
        }
        last = ri[0];
        __syncthreads();
        if (tid == 0) sel[it] = last;
    }
    __syncthreads();
    for (int s = tid; s < NPOINT_; s += 512) {
        int p = sel[s];
        fps_idx[b * NPOINT_ + s] = p;
        size_t o = ((size_t)b * NPOINT_ + s) * 3;
        new_xyz[o + 0] = xb[p * 3 + 0];
        new_xyz[o + 1] = xb[p * 3 + 1];
        new_xyz[o + 2] = xb[p * 3 + 2];
    }
}

// ---------------------------------------------------------------------------
// kNN (top-32): one wave per query; distances in LDS; 32 rounds of wave-wide
// argmin via __shfl_xor (wave32), tie -> lower index.
// ---------------------------------------------------------------------------
__global__ void knn_kernel(const float* __restrict__ xyz, const float* __restrict__ new_xyz,
                           int* __restrict__ knn_idx) {
    __shared__ float dist[NPTS];        // 32 KB
    const int q    = blockIdx.x;
    const int b    = q >> 11;
    const int lane = threadIdx.x;
    const float* xb = xyz + (size_t)b * NPTS * 3;
    const float qx = new_xyz[(size_t)q * 3 + 0];
    const float qy = new_xyz[(size_t)q * 3 + 1];
    const float qz = new_xyz[(size_t)q * 3 + 2];

    for (int i = lane; i < NPTS; i += 32) {
        float dx = xb[i * 3 + 0] - qx;
        float dy = xb[i * 3 + 1] - qy;
        float dz = xb[i * 3 + 2] - qz;
        dist[i] = dx * dx + dy * dy + dz * dz;
    }
    __syncthreads();

    for (int r = 0; r < NSAMPLE_; ++r) {
        float bv = 3.0e38f; int bi = NPTS;
        for (int i = lane; i < NPTS; i += 32) {
            float d = dist[i];
            if (d < bv) { bv = d; bi = i; }
        }
        #pragma unroll
        for (int off = 16; off > 0; off >>= 1) {
            float ov = __shfl_xor(bv, off, 32);
            int   oi = __shfl_xor(bi, off, 32);
            if (ov < bv || (ov == bv && oi < bi)) { bv = ov; bi = oi; }
        }
        if (lane == 0) { knn_idx[(size_t)q * NSAMPLE_ + r] = bi; dist[bi] = 3.0e38f; }
        __syncthreads();
    }
}

// ---------------------------------------------------------------------------
// Gather + concat(rel_xyz, feats) -> f16 rows written directly in A-fragment
// order (4 contiguous 32B vector stores per row).
// ---------------------------------------------------------------------------
__global__ void build_feat_kernel(const float* __restrict__ xyz, const float* __restrict__ points,
                                  const float* __restrict__ new_xyz, const int* __restrict__ knn,
                                  _Float16* __restrict__ feat) {
    int r = blockIdx.x * blockDim.x + threadIdx.x;
    if (r >= MROWS) return;
    int b   = r >> 16;
    int rem = r & 65535;
    int s   = rem >> 5;
    int pidx = knn[r];
    const float* xb = xyz + (size_t)b * NPTS * 3;
    size_t qo = ((size_t)b * NPOINT_ + s) * 3;

    float fv[KDIM];
    fv[0] = xb[pidx * 3 + 0] - new_xyz[qo + 0];
    fv[1] = xb[pidx * 3 + 1] - new_xyz[qo + 1];
    fv[2] = xb[pidx * 3 + 2] - new_xyz[qo + 2];
    const float* pb = points + ((size_t)b * NPTS + pidx) * DFEAT;
    #pragma unroll
    for (int c = 0; c < DFEAT; ++c) fv[3 + c] = pb[c];
    #pragma unroll
    for (int c = 3 + DFEAT; c < KDIM; ++c) fv[c] = 0.0f;

    const int mtile = r >> 4;
    const int ml    = r & 15;
    v16h* fo = (v16h*)feat;
    #pragma unroll
    for (int ks = 0; ks < 2; ++ks) {
        #pragma unroll
        for (int h = 0; h < 2; ++h) {
            v16h o;
            #pragma unroll
            for (int i = 0; i < 16; ++i) o[i] = (_Float16)fv[afrag_K(h, ks, i)];
            fo[(mtile * 2 + ks) * 32 + h * 16 + ml] = o;
        }
    }
}

// ---------------------------------------------------------------------------
// WMMA MLP layer: Out[M,NOUT] = act(A[M,64]) x W[64,NOUT]; f16 in / f32 acc.
// A and W arrive pre-packed in fragment order -> pure b128 coalesced loads.
// Epilogue: BN sum/sumsq from f32 accumulators, LDS transpose of the
// column-striped D tile into row-striped A-fragment order for the next layer.
// Block = 128 threads = 4 waves; each wave owns one 16-row M tile.
// ---------------------------------------------------------------------------
template <int NOUT, bool AFFINE>
__global__ __launch_bounds__(128) void mlp_wmma(const _Float16* __restrict__ A,
                                                const _Float16* __restrict__ W,
                                                const float* __restrict__ aff,   // [64]=scale,[64..]=shift
                                                float* __restrict__ gsum, float* __restrict__ gsq,
                                                _Float16* __restrict__ Out) {
    __shared__ float    ls[NOUT];
    __shared__ float    lq[NOUT];
    __shared__ _Float16 tile[4][16 * NOUT];   // per-wave 16 x NOUT f16 staging
    const int tid = threadIdx.x;
    for (int c = tid; c < NOUT; c += 128) { ls[c] = 0.0f; lq[c] = 0.0f; }
    __syncthreads();

    const int wave  = tid >> 5;
    const int lane  = tid & 31;
    const int half  = lane >> 4;
    const int ml    = lane & 15;
    const int mtile = blockIdx.x * 4 + wave;

    const v16h* Av = (const v16h*)A;
    const v16h* Wv = (const v16h*)W;

    v16h afrag[2];
    #pragma unroll
    for (int ks = 0; ks < 2; ++ks) {
        afrag[ks] = Av[(mtile * 2 + ks) * 32 + lane];
        if (AFFINE) {
            #pragma unroll
            for (int i = 0; i < 16; ++i) {
                int K = afrag_K(half, ks, i);
                float v = (float)afrag[ks][i];
                afrag[ks][i] = (_Float16)fmaxf(fmaf(aff[K], v, aff[KDIM + K]), 0.0f);
            }
        }
    }

    #pragma unroll
    for (int nt = 0; nt < NOUT / 16; ++nt) {
        v8f acc;
        #pragma unroll
        for (int i = 0; i < 8; ++i) acc[i] = 0.0f;
        #pragma unroll
        for (int ks = 0; ks < 2; ++ks) {
            v16h bfrag = Wv[(nt * 2 + ks) * 32 + lane];
            acc = __builtin_amdgcn_wmma_f32_16x16x32_f16(false, afrag[ks], false, bfrag,
                                                         (short)0, acc, false, false);
        }
        float s = 0.0f, q = 0.0f;
        #pragma unroll
        for (int r = 0; r < 8; ++r) {     // D layout: VGPR r -> row = half*8+r, col = nt*16+ml
            float v = acc[r];
            tile[wave][(half * 8 + r) * NOUT + nt * 16 + ml] = (_Float16)v;
            s += v; q += v * v;
        }
        atomicAdd(&ls[nt * 16 + ml], s);
        atomicAdd(&lq[nt * 16 + ml], q);
    }
    __syncthreads();

    // Re-emit the tile as next-layer A fragments: contiguous 32B per lane.
    v16h* Ov = (v16h*)Out;
    #pragma unroll
    for (int ks = 0; ks < NOUT / 32; ++ks) {
        v16h o;
        #pragma unroll
        for (int i = 0; i < 16; ++i) o[i] = tile[wave][ml * NOUT + afrag_K(half, ks, i)];
        Ov[(mtile * (NOUT / 32) + ks) * 32 + lane] = o;
    }

    for (int c = tid; c < NOUT; c += 128) {
        atomicAdd(&gsum[c], ls[c]);
        atomicAdd(&gsq[c], lq[c]);
    }
}

// BN finalize: scale = g*rsqrt(var+eps); shift = be - mu*scale.
// (Conv bias cancels exactly under mean-subtracting BN, so it is skipped.)
__global__ void finalize_kernel(const float* __restrict__ gsum, const float* __restrict__ gsq,
                                const float* __restrict__ g, const float* __restrict__ be,
                                float* __restrict__ aff, int nout) {
    int c = threadIdx.x;
    if (c >= nout) return;
    const float inv = 1.0f / (float)MROWS;
    float mu  = gsum[c] * inv;
    float var = gsq[c] * inv - mu * mu;
    float a   = g[c] * rsqrtf(var + BN_EPS_);
    aff[c] = a;
    aff[nout + c] = be[c] - mu * a;
}

// Apply BN3+ReLU and max over K=32 neighbors; act3 is in fragment order (C=128).
__global__ void maxpool_kernel(const _Float16* __restrict__ act3, const float* __restrict__ aff3,
                               float* __restrict__ out) {
    const int bs = blockIdx.x;          // 0..B*S-1
    const int c  = threadIdx.x;         // 0..127
    const float a = aff3[c], bb = aff3[128 + c];
    const int ks   = c >> 5;
    const int rem  = c & 31;
    const int half = (rem >> 3) & 1;
    const int i    = ((rem >> 4) << 3) + (rem & 7);
    float m = 0.0f;                     // relu() >= 0
    #pragma unroll
    for (int k = 0; k < NSAMPLE_; ++k) {
        int row   = bs * NSAMPLE_ + k;
        int mtile = row >> 4;
        int lane  = half * 16 + (row & 15);
        float x = (float)act3[(size_t)(((mtile * 4 + ks) * 32 + lane) * 16 + i)];
        m = fmaxf(m, fmaxf(fmaf(a, x, bb), 0.0f));
    }
    out[(size_t)bs * 128 + c] = m;
}

// ---------------------------------------------------------------------------
extern "C" void kernel_launch(void* const* d_in, const int* in_sizes, int n_in,
                              void* d_out, int out_size, void* d_ws, size_t ws_size,
                              hipStream_t stream) {
    (void)in_sizes; (void)n_in; (void)out_size; (void)ws_size;
    const float* xyz    = (const float*)d_in[0];
    const float* points = (const float*)d_in[1];
    const float* w1 = (const float*)d_in[2];
    const float* g1 = (const float*)d_in[4];
    const float* be1= (const float*)d_in[5];
    const float* w2 = (const float*)d_in[6];
    const float* g2 = (const float*)d_in[8];
    const float* be2= (const float*)d_in[9];
    const float* w3 = (const float*)d_in[10];
    const float* g3 = (const float*)d_in[12];
    const float* be3= (const float*)d_in[13];

    float* out        = (float*)d_out;
    float* new_xyz    = out;                                  // [8,2048,3]
    float* new_points = out + (size_t)BATCH * NPOINT_ * 3;    // [8,2048,128]

    char* ws = (char*)d_ws;
    _Float16* regA = (_Float16*)(ws);                         // 64 MB: feat, then act2
    _Float16* regB = (_Float16*)(ws + (64ull << 20));         // 128 MB: act1, then act3
    int* knn    = (int*)(ws + (192ull << 20));                // 2 MB
    int* fpsIdx = (int*)(ws + (194ull << 20));                // 64 KB
    _Float16* wf1 = (_Float16*)(ws + (194ull << 20) + (1u << 16));
    _Float16* wf2 = wf1 + KDIM * 64;
    _Float16* wf3 = wf2 + KDIM * 64;
    float* stats = (float*)(wf3 + KDIM * 128);                // 512 floats of sums
    float* gs1 = stats,       *gq1 = stats + 64;
    float* gs2 = stats + 128, *gq2 = stats + 192;
    float* gs3 = stats + 256, *gq3 = stats + 384;
    float* aff1 = stats + 512;          // 128
    float* aff2 = aff1 + 128;           // 128
    float* aff3 = aff2 + 128;           // 256

    prep_kernel<<<32, 256, 0, stream>>>(w1, w2, w3, wf1, wf2, wf3, stats);
    fps_kernel<<<BATCH, 512, 0, stream>>>(xyz, fpsIdx, new_xyz);
    knn_kernel<<<BATCH * NPOINT_, 32, 0, stream>>>(xyz, new_xyz, knn);
    build_feat_kernel<<<MROWS / 256, 256, 0, stream>>>(xyz, points, new_xyz, knn, regA);

    mlp_wmma<64, false><<<MROWS / 64, 128, 0, stream>>>(regA, wf1, nullptr, gs1, gq1, regB);
    finalize_kernel<<<1, 64, 0, stream>>>(gs1, gq1, g1, be1, aff1, 64);
    mlp_wmma<64, true><<<MROWS / 64, 128, 0, stream>>>(regB, wf2, aff1, gs2, gq2, regA);
    finalize_kernel<<<1, 64, 0, stream>>>(gs2, gq2, g2, be2, aff2, 64);
    mlp_wmma<128, true><<<MROWS / 64, 128, 0, stream>>>(regA, wf3, aff2, gs3, gq3, regB);
    finalize_kernel<<<1, 128, 0, stream>>>(gs3, gq3, g3, be3, aff3, 128);

    maxpool_kernel<<<BATCH * NPOINT_, 128, 0, stream>>>(regB, aff3, new_points);
}
